// Block_1408749273889
// MI455X (gfx1250) — matrix operand order repaired and checked by reading, not compiled, not run on hardware
//
#include <hip/hip_runtime.h>
#include <cstdint>
#include <cstddef>

// ---------------------------------------------------------------------------
// Types for CDNA5 WMMA (wave32): v_wmma_f32_16x16x32_f16
// ---------------------------------------------------------------------------
typedef __attribute__((ext_vector_type(16))) _Float16 v16h;
typedef __attribute__((ext_vector_type(8)))  _Float16 v8h;
typedef __attribute__((ext_vector_type(4)))  _Float16 v4h;
typedef __attribute__((ext_vector_type(8)))  float    v8f;

#define N_EMBD 1024
#define N_HEAD 16
#define HD     64
#define SEQ    2048
#define BT     4096   // B*T rows

// ---------------------------------------------------------------------------
// Fragment loaders (layouts per CDNA5 ISA 7.12.2)
// A 16x32 f16: lane = {row = l&15, hf = l>>4}; elems 0..7 -> K = hf*8+i,
//              elems 8..15 -> K = 16 + hf*8 + (i-8). Two b128 LDS loads.
// B 32x16 f16 from "Bt" layout (tile stored [N][K] row-major): lane col = l&15,
//              elems 0..7 -> K = hf*8+i, elems 8..15 -> K = 16 + hf*8 + (i-8). Two b128 LDS loads.
// C/D 16x16 f32: VGPR r -> row = r + 8*hf, col = l&15.
// ---------------------------------------------------------------------------
__device__ __forceinline__ v16h frag_a(const _Float16* base, int stride, int lane) {
  const int row = lane & 15, hf = lane >> 4;
  const _Float16* p = base + row * stride;
  v8h lo = *(const v8h*)(p + hf * 8);
  v8h hi = *(const v8h*)(p + 16 + hf * 8);
  v16h r;
#pragma unroll
  for (int i = 0; i < 8; ++i) { r[i] = lo[i]; r[i + 8] = hi[i]; }
  return r;
}

__device__ __forceinline__ v16h frag_b(const _Float16* base, int stride, int lane) {
  const int col = lane & 15, hf = lane >> 4;
  const _Float16* p = base + col * stride + hf * 8;
  v8h lo = *(const v8h*)(p);
  v8h hi = *(const v8h*)(p + 16);
  v16h r;
#pragma unroll
  for (int i = 0; i < 8; ++i) { r[i] = lo[i]; r[i + 8] = hi[i]; }
  return r;
}

__device__ __forceinline__ v8f wmma32(v16h a, v16h b, v8f c) {
  return __builtin_amdgcn_wmma_f32_16x16x32_f16(false, a, false, b, (short)0, c,
                                                false, false);
}

__device__ __forceinline__ v8f vzero8() {
  v8f z;
#pragma unroll
  for (int i = 0; i < 8; ++i) z[i] = 0.0f;
  return z;
}

// ---------------------------------------------------------------------------
// f32 -> f16 weight conversion
// ---------------------------------------------------------------------------
__global__ void cvt_f16_kernel(const float* __restrict__ src,
                               _Float16* __restrict__ dst, int n) {
  int i = blockIdx.x * blockDim.x + threadIdx.x;
  if (i < n) dst[i] = (_Float16)src[i];
}

// ---------------------------------------------------------------------------
// LayerNorm: one row (1024) per 256-thread block, f32 in -> f16 out
// ---------------------------------------------------------------------------
__global__ __launch_bounds__(256) void ln_kernel(const float* __restrict__ x,
                                                 const float* __restrict__ g,
                                                 const float* __restrict__ b,
                                                 _Float16* __restrict__ out) {
  const int row = blockIdx.x;
  const int tid = threadIdx.x;
  const int lane = tid & 31, wid = tid >> 5;
  const float* xr = x + (size_t)row * N_EMBD;

  float4 v = ((const float4*)xr)[tid];
  float s  = v.x + v.y + v.z + v.w;
  float sq = v.x * v.x + v.y * v.y + v.z * v.z + v.w * v.w;
#pragma unroll
  for (int m = 1; m < 32; m <<= 1) {
    s  += __shfl_xor(s, m, 32);
    sq += __shfl_xor(sq, m, 32);
  }
  __shared__ float sh[16];
  if (lane == 0) { sh[wid] = s; sh[8 + wid] = sq; }
  __syncthreads();
  float S = 0.f, Q = 0.f;
#pragma unroll
  for (int i = 0; i < 8; ++i) { S += sh[i]; Q += sh[8 + i]; }
  const float mean = S * (1.0f / N_EMBD);
  const float var  = Q * (1.0f / N_EMBD) - mean * mean;
  const float rstd = rsqrtf(var + 1e-5f);

  float4 gv = ((const float4*)g)[tid];
  float4 bv = ((const float4*)b)[tid];
  v4h o;
  o[0] = (_Float16)((v.x - mean) * rstd * gv.x + bv.x);
  o[1] = (_Float16)((v.y - mean) * rstd * gv.y + bv.y);
  o[2] = (_Float16)((v.z - mean) * rstd * gv.z + bv.z);
  o[3] = (_Float16)((v.w - mean) * rstd * gv.w + bv.w);
  ((v4h*)(out + (size_t)row * N_EMBD))[tid] = o;
}

// ---------------------------------------------------------------------------
// Generic WMMA GEMM, software-pipelined with ping-pong LDS buffers.
// 256 threads = 8 waves; tile 128x128, BK=32; wave -> 64x32 (4x2 frags).
// Stage s+1 is fetched to registers while stage s computes; one barrier/stage.
// MODE 0: QKV scatter (q,k f16 [B,H,T,64]; v f16 transposed [B,H,64,T])
// MODE 1: proj: f32 out = acc + bias + resid
// MODE 2: fc:   f16 out = gelu_exact(acc + bias)
// MODE 3: fc2:  f32 out = acc + bias + resid
// ---------------------------------------------------------------------------
#define BM 128
#define BN 128
#define BK 32
#define LPAD 8
#define LSTR (BK + LPAD)

template <int MODE>
__global__ __launch_bounds__(256) void gemm_kernel(
    const _Float16* __restrict__ A, const _Float16* __restrict__ B,
    const float* __restrict__ bias, const float* __restrict__ resid,
    float* __restrict__ outF, _Float16* __restrict__ outH,
    _Float16* __restrict__ qp, _Float16* __restrict__ kp,
    _Float16* __restrict__ vtp, int M, int N, int K) {
  __shared__ _Float16 As[2][BM][LSTR];
  __shared__ _Float16 Bt[2][BN][LSTR];

  const int tid = threadIdx.x;
  const int lane = tid & 31, wid = tid >> 5;
  const int m0 = blockIdx.y * BM, n0 = blockIdx.x * BN;
  const int wm = (wid >> 2) * 64;  // 0 or 64
  const int wn = (wid & 3) * 32;   // 0,32,64,96

  v8f acc[4][2];
#pragma unroll
  for (int i = 0; i < 4; ++i)
#pragma unroll
    for (int j = 0; j < 2; ++j) acc[i][j] = vzero8();

  const int arow = tid >> 1, aseg = tid & 1;  // A: 2 threads/row, 16 f16 each
  const int krow = tid >> 3, nseg = tid & 7;  // B: 8 threads/k-row, 16 f16 each
  const _Float16* Ag = A + (size_t)(m0 + arow) * K + aseg * 16;
  const _Float16* Bg = B + (size_t)krow * N + n0 + nseg * 16;

  auto stage_store = [&](int p, uint4 sa0, uint4 sa1, uint4 sb0, uint4 sb1) {
    *(uint4*)&As[p][arow][aseg * 16]     = sa0;
    *(uint4*)&As[p][arow][aseg * 16 + 8] = sa1;
    const _Float16* h0 = (const _Float16*)&sb0;
    const _Float16* h1 = (const _Float16*)&sb1;
#pragma unroll
    for (int i = 0; i < 8; ++i) Bt[p][nseg * 16 + i][krow] = h0[i];
#pragma unroll
    for (int i = 0; i < 8; ++i) Bt[p][nseg * 16 + 8 + i][krow] = h1[i];
  };

  auto compute = [&](int p) {
    v16h bf[2], af[4];
#pragma unroll
    for (int j = 0; j < 2; ++j) bf[j] = frag_b(&Bt[p][wn + j * 16][0], LSTR, lane);
#pragma unroll
    for (int i = 0; i < 4; ++i) af[i] = frag_a(&As[p][wm + i * 16][0], LSTR, lane);
#pragma unroll
    for (int i = 0; i < 4; ++i)
#pragma unroll
      for (int j = 0; j < 2; ++j) acc[i][j] = wmma32(af[i], bf[j], acc[i][j]);
  };

  // prologue: fetch+stage chunk 0
  {
    uint4 fa0 = *(const uint4*)(Ag);
    uint4 fa1 = *(const uint4*)(Ag + 8);
    uint4 fb0 = *(const uint4*)(Bg);
    uint4 fb1 = *(const uint4*)(Bg + 8);
    stage_store(0, fa0, fa1, fb0, fb1);
  }

  const int nst = K / BK;
  int p = 0;
  for (int s = 1; s < nst; ++s) {
    // issue global fetch of stage s (latency hidden behind compute of stage s-1)
    const _Float16* ap = Ag + s * BK;
    const _Float16* bp = Bg + (size_t)s * BK * N;
    uint4 fa0 = *(const uint4*)(ap);
    uint4 fa1 = *(const uint4*)(ap + 8);
    uint4 fb0 = *(const uint4*)(bp);
    uint4 fb1 = *(const uint4*)(bp + 8);
    __syncthreads();           // stage s-1 LDS visible
    compute(p);
    stage_store(p ^ 1, fa0, fa1, fb0, fb1);
    p ^= 1;
  }
  __syncthreads();
  compute(p);

  // epilogue
  const int nlane = lane & 15, hf = lane >> 4;
#pragma unroll
  for (int i = 0; i < 4; ++i) {
#pragma unroll
    for (int j = 0; j < 2; ++j) {
      const int col = n0 + wn + j * 16 + nlane;
      const float bv = bias[col];
#pragma unroll
      for (int r = 0; r < 8; ++r) {
        const int row = m0 + wm + i * 16 + hf * 8 + r;
        float val = acc[i][j][r] + bv;
        if constexpr (MODE == 0) {
          const int part = col >> 10;  // 0=q 1=k 2=v
          const int cc = col & 1023;
          const int hh = cc >> 6, dd = cc & 63;
          const int bb = row >> 11, t = row & 2047;
          const size_t bh = (size_t)(bb * N_HEAD + hh);
          if (part == 0)      qp[(bh * SEQ + t) * HD + dd] = (_Float16)val;
          else if (part == 1) kp[(bh * SEQ + t) * HD + dd] = (_Float16)val;
          else                vtp[(bh * HD + dd) * SEQ + t] = (_Float16)val;
        } else if constexpr (MODE == 1) {
          const size_t idx = (size_t)row * N + col;
          outF[idx] = val + resid[idx];
        } else if constexpr (MODE == 2) {
          const float ge = 0.5f * val * (1.0f + erff(val * 0.70710678118654752f));
          outH[(size_t)row * N + col] = (_Float16)ge;
        } else {
          const size_t idx = (size_t)row * N + col;
          outF[idx] = val + resid[idx];
        }
      }
    }
  }
}

// ---------------------------------------------------------------------------
// Causal flash attention, pipelined over key blocks with ping-pong K/V^T LDS.
// Grid: (T/64, B*H). 128 threads = 4 waves; wave w owns q-rows [16w,16w+16).
// q tile [64][64]; K block row-major (== Bt layout for Q*K^T);
// V^T block [64 d][64 keys] (== Bt layout for P*V); per-wave P [16][64].
// ---------------------------------------------------------------------------
#define AST 72  // padded LDS row stride (f16)

__global__ __launch_bounds__(128) void attn_kernel(
    const _Float16* __restrict__ q, const _Float16* __restrict__ k,
    const _Float16* __restrict__ vt, _Float16* __restrict__ y) {
  __shared__ _Float16 qs[64][AST];
  __shared__ _Float16 ks[2][64][AST];
  __shared__ _Float16 vts[2][64][AST];
  __shared__ _Float16 ps[4][16][AST];

  const int tid = threadIdx.x;
  const int lane = tid & 31, wid = tid >> 5;
  const int nlane = lane & 15, hf = lane >> 4;
  const int bh = blockIdx.y;
  const int bb = bh >> 4, hh = bh & 15;
  const int q0 = blockIdx.x * 64;

  const int lrowi = tid >> 1, lseg = tid & 1;  // tile loads: 2 threads/row
  const _Float16* kgb = k + ((size_t)bh * SEQ + lrowi) * HD + lseg * 32;
  const _Float16* vgb = vt + ((size_t)bh * HD + lrowi) * SEQ + lseg * 32;

  // stage Q tile once
  {
    const uint4* g = (const uint4*)(q + ((size_t)bh * SEQ + q0 + lrowi) * HD + lseg * 32);
#pragma unroll
    for (int i = 0; i < 4; ++i) *(uint4*)&qs[lrowi][lseg * 32 + i * 8] = g[i];
  }

  float mrow[8], lrow[8];
  v8f o[4];
#pragma unroll
  for (int r = 0; r < 8; ++r) { mrow[r] = -1e30f; lrow[r] = 0.0f; }
#pragma unroll
  for (int nt = 0; nt < 4; ++nt) o[nt] = vzero8();

  // prologue: stage key-block 0
  {
    const uint4* gk = (const uint4*)(kgb);
    const uint4* gv = (const uint4*)(vgb);
#pragma unroll
    for (int i = 0; i < 4; ++i) *(uint4*)&ks[0][lrowi][lseg * 32 + i * 8] = gk[i];
#pragma unroll
    for (int i = 0; i < 4; ++i) *(uint4*)&vts[0][lrowi][lseg * 32 + i * 8] = gv[i];
  }

  const int jmax = blockIdx.x;  // key blocks 0..q0/64
  int p = 0;
  for (int j = 0; j <= jmax; ++j) {
    // issue global fetch of next key block (hidden behind this block's math)
    uint4 nk[4], nv[4];
    if (j < jmax) {
      const int kb1 = (j + 1) * 64;
      const uint4* gk = (const uint4*)(kgb + (size_t)kb1 * HD);
      const uint4* gv = (const uint4*)(vgb + kb1);
#pragma unroll
      for (int i = 0; i < 4; ++i) { nk[i] = gk[i]; nv[i] = gv[i]; }
    }
    __syncthreads();  // (A) ks[p]/vts[p] (and qs on j==0) visible

    // S = Q * K^T (per wave: 16 x 64), scale + causal mask
    const int kb = j * 64;
    v16h aq[2];
#pragma unroll
    for (int ksn = 0; ksn < 2; ++ksn)
      aq[ksn] = frag_a(&qs[wid * 16][ksn * 32], AST, lane);
    float sv[4][8];
#pragma unroll
    for (int ct = 0; ct < 4; ++ct) {
      v8f s = vzero8();
#pragma unroll
      for (int ksn = 0; ksn < 2; ++ksn)
        s = wmma32(aq[ksn], frag_b(&ks[p][ct * 16][ksn * 32], AST, lane), s);
      const int kcol = kb + ct * 16 + nlane;
#pragma unroll
      for (int r = 0; r < 8; ++r) {
        const int qi = q0 + wid * 16 + hf * 8 + r;
        float t = s[r] * 0.125f;  // 1/sqrt(64)
        sv[ct][r] = (kcol > qi) ? -1e30f : t;
      }
    }

    // online softmax: per-row max/sum across 64 keys
#pragma unroll
    for (int r = 0; r < 8; ++r) {
      float pm = fmaxf(fmaxf(sv[0][r], sv[1][r]), fmaxf(sv[2][r], sv[3][r]));
#pragma unroll
      for (int m = 1; m < 16; m <<= 1) pm = fmaxf(pm, __shfl_xor(pm, m, 32));
      const float mnew = fmaxf(mrow[r], pm);
      const float corr = __expf(mrow[r] - mnew);
      mrow[r] = mnew;
      float rs = 0.0f;
#pragma unroll
      for (int ct = 0; ct < 4; ++ct) {
        const float pe = __expf(sv[ct][r] - mnew);
        rs += pe;
        ps[wid][hf * 8 + r][ct * 16 + nlane] = (_Float16)pe;
      }
#pragma unroll
      for (int m = 1; m < 16; m <<= 1) rs += __shfl_xor(rs, m, 32);
      lrow[r] = lrow[r] * corr + rs;
#pragma unroll
      for (int nt = 0; nt < 4; ++nt) o[nt][r] *= corr;
    }
    __syncthreads();  // (B) order ps stores before fragment reads

    // deposit prefetched next key block into the other buffer
    if (j < jmax) {
#pragma unroll
      for (int i = 0; i < 4; ++i)
        *(uint4*)&ks[p ^ 1][lrowi][lseg * 32 + i * 8] = nk[i];
#pragma unroll
      for (int i = 0; i < 4; ++i)
        *(uint4*)&vts[p ^ 1][lrowi][lseg * 32 + i * 8] = nv[i];
    }

    // O += P * V (per wave: 16 x 64)
    v16h ap[2];
#pragma unroll
    for (int ksn = 0; ksn < 2; ++ksn)
      ap[ksn] = frag_a(&ps[wid][0][ksn * 32], AST, lane);
#pragma unroll
    for (int nt = 0; nt < 4; ++nt)
#pragma unroll
      for (int ksn = 0; ksn < 2; ++ksn)
        o[nt] = wmma32(ap[ksn], frag_b(&vts[p][nt * 16][ksn * 32], AST, lane), o[nt]);

    p ^= 1;
  }

  // normalize and write y as f16 [B*T, 1024] (cols = h*64 + d)
#pragma unroll
  for (int r = 0; r < 8; ++r) {
    const float inv = 1.0f / lrow[r];
    const int t = q0 + wid * 16 + hf * 8 + r;
    const size_t base = ((size_t)bb * SEQ + t) * N_EMBD + hh * HD;
#pragma unroll
    for (int nt = 0; nt < 4; ++nt)
      y[base + nt * 16 + nlane] = (_Float16)(o[nt][r] * inv);
  }
}

// ---------------------------------------------------------------------------
// Host launch
// ---------------------------------------------------------------------------
extern "C" void kernel_launch(void* const* d_in, const int* in_sizes, int n_in,
                              void* d_out, int out_size, void* d_ws, size_t ws_size,
                              hipStream_t stream) {
  const float* x      = (const float*)d_in[0];
  const float* W_attn = (const float*)d_in[1];
  const float* b_attn = (const float*)d_in[2];
  const float* W_proj = (const float*)d_in[3];
  const float* b_proj = (const float*)d_in[4];
  const float* W_fc   = (const float*)d_in[5];
  const float* b_fc   = (const float*)d_in[6];
  const float* W_fc2  = (const float*)d_in[7];
  const float* b_fc2  = (const float*)d_in[8];
  const float* ln1_g  = (const float*)d_in[9];
  const float* ln1_b  = (const float*)d_in[10];
  const float* ln2_g  = (const float*)d_in[11];
  const float* ln2_b  = (const float*)d_in[12];
  float* out = (float*)d_out;

  char* ws = (char*)d_ws;
  const size_t SZ8 = (size_t)BT * N_EMBD * sizeof(_Float16);  // 8 MB
  _Float16* wbuf = (_Float16*)(ws);                // reused f16 weights (max 8 MB)
  _Float16* a16  = (_Float16*)(ws + SZ8);          // LN output (f16)
  _Float16* q16  = (_Float16*)(ws + 2 * SZ8);
  _Float16* k16  = (_Float16*)(ws + 3 * SZ8);
  _Float16* vt16 = (_Float16*)(ws + 4 * SZ8);
  _Float16* y16  = (_Float16*)(ws + 5 * SZ8);
  float*    x1   = (float*)(ws + 6 * SZ8);         // 16 MB
  _Float16* h16  = (_Float16*)(ws + 6 * SZ8 + (size_t)BT * N_EMBD * 4);  // 32 MB

  // ---- LN1 + QKV
  ln_kernel<<<BT, 256, 0, stream>>>(x, ln1_g, ln1_b, a16);
  {
    int n = N_EMBD * 3 * N_EMBD;
    cvt_f16_kernel<<<n / 256, 256, 0, stream>>>(W_attn, wbuf, n);
  }
  gemm_kernel<0><<<dim3(3 * N_EMBD / BN, BT / BM), 256, 0, stream>>>(
      a16, wbuf, b_attn, nullptr, nullptr, nullptr, q16, k16, vt16,
      BT, 3 * N_EMBD, N_EMBD);

  // ---- attention
  attn_kernel<<<dim3(SEQ / 64, 2 * N_HEAD), 128, 0, stream>>>(q16, k16, vt16, y16);

  // ---- proj + residual -> x1
  {
    int n = N_EMBD * N_EMBD;
    cvt_f16_kernel<<<n / 256, 256, 0, stream>>>(W_proj, wbuf, n);
  }
  gemm_kernel<1><<<dim3(N_EMBD / BN, BT / BM), 256, 0, stream>>>(
      y16, wbuf, b_proj, x, x1, nullptr, nullptr, nullptr, nullptr,
      BT, N_EMBD, N_EMBD);

  // ---- LN2 + FC + GELU
  ln_kernel<<<BT, 256, 0, stream>>>(x1, ln2_g, ln2_b, a16);
  {
    int n = N_EMBD * 4 * N_EMBD;
    cvt_f16_kernel<<<n / 256, 256, 0, stream>>>(W_fc, wbuf, n);
  }
  gemm_kernel<2><<<dim3(4 * N_EMBD / BN, BT / BM), 256, 0, stream>>>(
      a16, wbuf, b_fc, nullptr, nullptr, h16, nullptr, nullptr, nullptr,
      BT, 4 * N_EMBD, N_EMBD);

  // ---- FC2 + residual -> out
  {
    int n = 4 * N_EMBD * N_EMBD;
    cvt_f16_kernel<<<n / 256, 256, 0, stream>>>(W_fc2, wbuf, n);
  }
  gemm_kernel<3><<<dim3(N_EMBD / BN, BT / BM), 256, 0, stream>>>(
      h16, wbuf, b_fc2, x1, out, nullptr, nullptr, nullptr, nullptr,
      BT, N_EMBD, 4 * N_EMBD);
}
